// MultiHeadedAttention_28965259444509
// MI455X (gfx1250) — matrix operand order repaired and checked by reading, not compiled, not run on hardware
//
#include <hip/hip_runtime.h>
#include <hip/hip_bf16.h>

// MI455X / gfx1250 fused multi-head attention, bf16 WMMA pipeline.
// Stage 0: one-time f32->bf16 conversion of Wq/Wk/Wv/Wo (kills redundant VALU cvt in GEMMs)
// Stage 1: QKV projections (f32 acts x bf16 weights -> bf16 Q/K/V [B,H,S,DH], Q scaled 1/sqrt(DH))
// Stage 2: flash attention; K-chunk staged with GLOBAL_LOAD_ASYNC_TO_LDS_B128 (ASYNCcnt),
//          V transposed into LDS, online softmax, bf16 WMMA for QK^T and P*V
// Stage 3: output projection (bf16 ctx x bf16 Wo -> f32 out)

typedef __bf16 bf16;
typedef bf16  v16bf __attribute__((ext_vector_type(16)));
typedef bf16  v8bf  __attribute__((ext_vector_type(8)));
typedef float v8f   __attribute__((ext_vector_type(8)));

#define BB  2
#define SS  2048
#define DD  1024
#define HHN 16
#define DHH 64

static __device__ __forceinline__ v8f wmma_bf16(v16bf a, v16bf b, v8f c) {
  return __builtin_amdgcn_wmma_f32_16x16x32_bf16(false, a, false, b, (short)0, c, false, false);
}

static __device__ __forceinline__ v8f zero8() {
  v8f z = {0.f, 0.f, 0.f, 0.f, 0.f, 0.f, 0.f, 0.f};
  return z;
}

// A-fragment (16x32, bf16) from an f32 row. Lane layout (contiguous k, same order as the B loads):
// lane = 16*hh + r holds row r; elems 0..15 = K (k0+16*hh..k0+16*hh+15)
static __device__ __forceinline__ v16bf load_a_from_f32(const float* __restrict__ row, int k0, int hh) {
  const float4* p0 = (const float4*)(row + k0 + 16 * hh);
  const float4* p1 = (const float4*)(row + k0 + 16 * hh + 8);
  float4 a0 = p0[0], a1 = p0[1];
  float4 b0 = p1[0], b1 = p1[1];
  v16bf r;
  r[0] = (bf16)a0.x; r[1] = (bf16)a0.y; r[2]  = (bf16)a0.z; r[3]  = (bf16)a0.w;
  r[4] = (bf16)a1.x; r[5] = (bf16)a1.y; r[6]  = (bf16)a1.z; r[7]  = (bf16)a1.w;
  r[8] = (bf16)b0.x; r[9] = (bf16)b0.y; r[10] = (bf16)b0.z; r[11] = (bf16)b0.w;
  r[12]= (bf16)b1.x; r[13]= (bf16)b1.y; r[14] = (bf16)b1.z; r[15] = (bf16)b1.w;
  return r;
}

// A-fragment (16x32, bf16) from a bf16 row (two contiguous 16B runs).
static __device__ __forceinline__ v16bf load_a_bf16(const bf16* __restrict__ row, int k0, int hh) {
  v8bf lo = *(const v8bf*)(row + k0 + 16 * hh);
  v8bf hi = *(const v8bf*)(row + k0 + 16 * hh + 8);
  v16bf r;
#pragma unroll
  for (int i = 0; i < 8; ++i) { r[i] = lo[i]; r[8 + i] = hi[i]; }
  return r;
}

// B-fragment (32x16, bf16) from 16 contiguous bf16 (weight row / LDS row slice).
static __device__ __forceinline__ v16bf load_b16_contig(const bf16* p) {
  v8bf lo = *(const v8bf*)p;
  v8bf hi = *(const v8bf*)(p + 8);
  v16bf r;
#pragma unroll
  for (int i = 0; i < 8; ++i) { r[i] = lo[i]; r[8 + i] = hi[i]; }
  return r;
}

// ---------------------------------------------------------------------------
// Stage 0: one-time weight conversion f32 -> bf16.  grid = (512, 1, 4).
// Each thread converts 8 contiguous elements (2x b128 load -> 1x b128 store).
// ---------------------------------------------------------------------------
__global__ __launch_bounds__(256) void cvt_w_kernel(
    const float* __restrict__ Wq, const float* __restrict__ Wk,
    const float* __restrict__ Wv, const float* __restrict__ Wo,
    bf16* __restrict__ Wqb, bf16* __restrict__ Wkb,
    bf16* __restrict__ Wvb, bf16* __restrict__ Wob) {
  const float* src;
  bf16* dst;
  if (blockIdx.z == 0)      { src = Wq; dst = Wqb; }
  else if (blockIdx.z == 1) { src = Wk; dst = Wkb; }
  else if (blockIdx.z == 2) { src = Wv; dst = Wvb; }
  else                      { src = Wo; dst = Wob; }
  const size_t i0 = ((size_t)blockIdx.x * 256 + threadIdx.x) * 8;  // covers D*D = 1M
  float4 f0 = ((const float4*)(src + i0))[0];
  float4 f1 = ((const float4*)(src + i0))[1];
  v8bf o;
  o[0] = (bf16)f0.x; o[1] = (bf16)f0.y; o[2] = (bf16)f0.z; o[3] = (bf16)f0.w;
  o[4] = (bf16)f1.x; o[5] = (bf16)f1.y; o[6] = (bf16)f1.z; o[7] = (bf16)f1.w;
  *(v8bf*)(dst + i0) = o;
}

// ---------------------------------------------------------------------------
// Stage 1: fused QKV projection.  grid = (256, 1, 3); block = 256 (8 waves).
// Each wave computes a 32x64 strip of (x @ W.T + b): 2 A-frags reuse 4 B-frags
// (8 WMMAs / k-step).  Writes bf16 [B,H,S,DH]; Q pre-scaled by 1/sqrt(DH).
// ---------------------------------------------------------------------------
__global__ __launch_bounds__(256) void qkv_kernel(
    const float* __restrict__ xq, const float* __restrict__ xk, const float* __restrict__ xv,
    const bf16* __restrict__ Wqb, const float* __restrict__ bq,
    const bf16* __restrict__ Wkb, const float* __restrict__ bk,
    const bf16* __restrict__ Wvb, const float* __restrict__ bv,
    bf16* __restrict__ qb, bf16* __restrict__ kb, bf16* __restrict__ vb) {
  const int lane = threadIdx.x & 31;
  const int wave = threadIdx.x >> 5;
  const int r    = lane & 15;
  const int hh   = lane >> 4;

  const int tile = blockIdx.x * 8 + wave;     // 0..2047
  const int NT   = DD / 64;                   // 16 column strips
  const int m0   = (tile / NT) * 32;          // 128 row tiles of 32
  const int n0   = (tile % NT) * 64;

  const float* X;  const bf16* W;  const float* bias;  bf16* out;  float scale;
  if (blockIdx.z == 0)      { X = xq; W = Wqb; bias = bq; out = qb; scale = 0.125f; }
  else if (blockIdx.z == 1) { X = xk; W = Wkb; bias = bk; out = kb; scale = 1.0f;  }
  else                      { X = xv; W = Wvb; bias = bv; out = vb; scale = 1.0f;  }

  const float* arow0 = X + (size_t)(m0 + r) * DD;
  const float* arow1 = X + (size_t)(m0 + 16 + r) * DD;
  v8f acc[8] = {zero8(), zero8(), zero8(), zero8(), zero8(), zero8(), zero8(), zero8()};

  for (int k0 = 0; k0 < DD; k0 += 32) {
    v16bf a0 = load_a_from_f32(arow0, k0, hh);
    v16bf a1 = load_a_from_f32(arow1, k0, hh);
#pragma unroll
    for (int j = 0; j < 4; ++j) {
      v16bf b = load_b16_contig(W + (size_t)(n0 + 16 * j + r) * DD + k0 + 16 * hh);
      acc[j]     = wmma_bf16(a0, b, acc[j]);
      acc[4 + j] = wmma_bf16(a1, b, acc[4 + j]);
    }
  }

#pragma unroll
  for (int j = 0; j < 4; ++j) {
    const int   n    = n0 + 16 * j + r;
    const float bn   = bias[n];
    const int   head = n >> 6;
    const int   d    = n & 63;
#pragma unroll
    for (int g = 0; g < 2; ++g)
#pragma unroll
      for (int jj = 0; jj < 8; ++jj) {
        const int m    = m0 + 16 * g + jj + 8 * hh;   // C layout: VGPR jj -> row jj+8*hh
        const int bidx = m >> 11;                     // /S
        const int s    = m & (SS - 1);
        const float v  = (acc[4 * g + j][jj] + bn) * scale;
        out[(((size_t)bidx * HHN + head) * SS + s) * DHH + d] = (bf16)v;
      }
  }
}

// ---------------------------------------------------------------------------
// Stage 2: flash attention.  grid = (S/128, H, B); block = 256 (8 waves).
// Each wave owns 16 q rows; block shares K/V kv-chunks of 64 via LDS.
// K-chunk is staged by the async DMA path (GLOBAL_LOAD_ASYNC_TO_LDS_B128,
// tracked by ASYNCcnt); V is transposed into LDS through VGPRs.
// ---------------------------------------------------------------------------
__global__ __launch_bounds__(256) void attn_kernel(
    const bf16* __restrict__ qb, const bf16* __restrict__ kb, const bf16* __restrict__ vb,
    const unsigned char* __restrict__ mask, bf16* __restrict__ ctxb) {
  __shared__ __align__(16) bf16 sK[64][64];    // [kv][d]  async-staged, b128-aligned rows
  __shared__ __align__(16) bf16 sVt[64][72];   // [d][kv]  transposed at stage time (padded)
  __shared__ __align__(16) bf16 sP[8][16][64]; // per-wave P tile: C-layout -> A-layout reshape

  const int lane = threadIdx.x & 31;
  const int wave = threadIdx.x >> 5;
  const int r    = lane & 15;
  const int hh   = lane >> 4;
  const int b    = blockIdx.z;
  const int h    = blockIdx.y;
  const int q0   = blockIdx.x * 128 + wave * 16;

  const bf16* qbase = qb + ((size_t)b * HHN + h) * SS * DHH;
  const bf16* kbase = kb + ((size_t)b * HHN + h) * SS * DHH;
  const bf16* vbase = vb + ((size_t)b * HHN + h) * SS * DHH;
  const unsigned char* mbase = mask + (size_t)b * SS * SS;

  // Q A-fragments (K-dim = DH = 64 -> two 16x32 fragments), loaded once.
  const bf16* qrow = qbase + (size_t)(q0 + r) * DHH;
  const v16bf aQ0 = load_a_bf16(qrow, 0, hh);
  const v16bf aQ1 = load_a_bf16(qrow, 32, hh);

  float mrow[8], lrow[8];
#pragma unroll
  for (int jj = 0; jj < 8; ++jj) { mrow[jj] = -1e30f; lrow[jj] = 0.f; }
  v8f accd[4] = {zero8(), zero8(), zero8(), zero8()};

  for (int kv0 = 0; kv0 < SS; kv0 += 64) {
    // Cooperative stage: K via async global->LDS DMA, V transposed via VGPRs.
#pragma unroll
    for (int i = 0; i < 2; ++i) {
      const int g  = threadIdx.x + 256 * i;   // 0..511 -> 64 rows x 8 vec8
      const int kv = g >> 3;
      const int c0 = (g & 7) * 8;
      const bf16* ksrc = kbase + (size_t)(kv0 + kv) * DHH + c0;
      const unsigned int ldsa = (unsigned int)(uintptr_t)&sK[kv][c0];
      asm volatile("global_load_async_to_lds_b128 %0, %1, off"
                   :: "v"(ldsa), "v"((unsigned long long)(uintptr_t)ksrc)
                   : "memory");
      v8bf vvec = *(const v8bf*)(vbase + (size_t)(kv0 + kv) * DHH + c0);
#pragma unroll
      for (int e = 0; e < 8; ++e) sVt[c0 + e][kv] = vvec[e];
    }
    if (kv0 + 64 < SS) {  // prefetch next kv chunk (global_prefetch_b8)
      const int row = (int)(threadIdx.x >> 2);
      __builtin_prefetch(kbase + (size_t)(kv0 + 64 + row) * DHH, 0, 0);
      __builtin_prefetch(vbase + (size_t)(kv0 + 64 + row) * DHH, 0, 0);
    }
    asm volatile("s_wait_asynccnt 0x0" ::: "memory");  // K DMA complete (this wave)
    __syncthreads();

    // scores: 4 tiles of 16x16 over the 64 kv columns (2 WMMAs each, K=64).
    v8f sc[4];
#pragma unroll
    for (int t = 0; t < 4; ++t) {
      v16bf b0 = load_b16_contig(&sK[16 * t + r][16 * hh]);       // d = 0..31 half
      v16bf b1 = load_b16_contig(&sK[16 * t + r][32 + 16 * hh]);  // d = 32..63 half
      v8f c = zero8();
      c = wmma_bf16(aQ0, b0, c);
      c = wmma_bf16(aQ1, b1, c);
      sc[t] = c;
    }

    // mask + online softmax. Row m = jj + 8*hh lives in VGPR jj across a
    // 16-lane half -> butterfly shfl_xor 1/2/4/8 stays inside the half.
#pragma unroll
    for (int jj = 0; jj < 8; ++jj) {
      const int qi = q0 + jj + 8 * hh;
      const unsigned char* mp = mbase + (size_t)qi * SS + kv0 + r;
      float mx = -1e30f;
#pragma unroll
      for (int t = 0; t < 4; ++t) {
        float s = sc[t][jj];
        s = mp[16 * t] ? s : -1e9f;
        sc[t][jj] = s;
        mx = fmaxf(mx, s);
      }
      mx = fmaxf(mx, __shfl_xor(mx, 1));
      mx = fmaxf(mx, __shfl_xor(mx, 2));
      mx = fmaxf(mx, __shfl_xor(mx, 4));
      mx = fmaxf(mx, __shfl_xor(mx, 8));
      const float newm  = fmaxf(mrow[jj], mx);
      const float alpha = __expf(mrow[jj] - newm);
      float rs = 0.f;
#pragma unroll
      for (int t = 0; t < 4; ++t) {
        const float p = __expf(sc[t][jj] - newm);
        sc[t][jj] = p;
        rs += p;
      }
      rs += __shfl_xor(rs, 1);
      rs += __shfl_xor(rs, 2);
      rs += __shfl_xor(rs, 4);
      rs += __shfl_xor(rs, 8);
      lrow[jj] = lrow[jj] * alpha + rs;
      mrow[jj] = newm;
#pragma unroll
      for (int ns = 0; ns < 4; ++ns) accd[ns][jj] *= alpha;   // rescale ctx acc
    }

    // P: C-layout -> natural [row][col] in per-wave LDS slab (bf16).
#pragma unroll
    for (int t = 0; t < 4; ++t)
#pragma unroll
      for (int jj = 0; jj < 8; ++jj)
        sP[wave][jj + 8 * hh][16 * t + r] = (bf16)sc[t][jj];

    asm volatile("s_wait_dscnt 0x0" ::: "memory");  // wave-local LDS RAW fence

    // ctx += P (16x64) x V (64x64): 2 k-steps x 4 d-strips.
#pragma unroll
    for (int ks = 0; ks < 2; ++ks) {
      v16bf aP = load_a_bf16(&sP[wave][r][0], ks * 32, hh);
#pragma unroll
      for (int ns = 0; ns < 4; ++ns) {
        v16bf bV = load_b16_contig(&sVt[16 * ns + r][ks * 32 + 16 * hh]);
        accd[ns] = wmma_bf16(aP, bV, accd[ns]);
      }
    }
    __syncthreads();
  }

  // Normalize and write ctx in merged-head [B,S,D] bf16 layout.
#pragma unroll
  for (int ns = 0; ns < 4; ++ns) {
    const int d = 16 * ns + r;
#pragma unroll
    for (int jj = 0; jj < 8; ++jj) {
      const int q   = q0 + jj + 8 * hh;
      const float v = accd[ns][jj] / lrow[jj];
      ctxb[((size_t)b * SS + q) * DD + h * DHH + d] = (bf16)v;
    }
  }
}

// ---------------------------------------------------------------------------
// Stage 3: output projection.  grid = (256); block = 256 (8 waves).
// out = ctx @ Wo.T + bo   (bf16 A, bf16 weights, f32 out), 32x64 wave tiles.
// ---------------------------------------------------------------------------
__global__ __launch_bounds__(256) void proj_kernel(
    const bf16* __restrict__ ctxb, const bf16* __restrict__ Wob,
    const float* __restrict__ bo, float* __restrict__ out) {
  const int lane = threadIdx.x & 31;
  const int wave = threadIdx.x >> 5;
  const int r    = lane & 15;
  const int hh   = lane >> 4;

  const int tile = blockIdx.x * 8 + wave;     // 0..2047
  const int NT   = DD / 64;
  const int m0   = (tile / NT) * 32;
  const int n0   = (tile % NT) * 64;

  const bf16* arow0 = ctxb + (size_t)(m0 + r) * DD;
  const bf16* arow1 = ctxb + (size_t)(m0 + 16 + r) * DD;
  v8f acc[8] = {zero8(), zero8(), zero8(), zero8(), zero8(), zero8(), zero8(), zero8()};

  for (int k0 = 0; k0 < DD; k0 += 32) {
    v16bf a0 = load_a_bf16(arow0, k0, hh);
    v16bf a1 = load_a_bf16(arow1, k0, hh);
#pragma unroll
    for (int j = 0; j < 4; ++j) {
      v16bf b = load_b16_contig(Wob + (size_t)(n0 + 16 * j + r) * DD + k0 + 16 * hh);
      acc[j]     = wmma_bf16(a0, b, acc[j]);
      acc[4 + j] = wmma_bf16(a1, b, acc[4 + j]);
    }
  }

#pragma unroll
  for (int j = 0; j < 4; ++j) {
    const int   n  = n0 + 16 * j + r;
    const float bn = bo[n];
#pragma unroll
    for (int g = 0; g < 2; ++g)
#pragma unroll
      for (int jj = 0; jj < 8; ++jj) {
        const int m = m0 + 16 * g + jj + 8 * hh;
        out[(size_t)m * DD + n] = acc[4 * g + j][jj] + bn;
      }
  }
}

extern "C" void kernel_launch(void* const* d_in, const int* in_sizes, int n_in,
                              void* d_out, int out_size, void* d_ws, size_t ws_size,
                              hipStream_t stream) {
  (void)in_sizes; (void)n_in; (void)out_size; (void)ws_size;
  const float* attn_from = (const float*)d_in[0];
  const float* attn_to   = (const float*)d_in[1];
  const float* value     = (const float*)d_in[2];
  const unsigned char* mask = (const unsigned char*)d_in[3];  // jnp bool -> 1 byte
  const float* Wq = (const float*)d_in[4];
  const float* bq = (const float*)d_in[5];
  const float* Wk = (const float*)d_in[6];
  const float* bk = (const float*)d_in[7];
  const float* Wv = (const float*)d_in[8];
  const float* bv = (const float*)d_in[9];
  const float* Wo = (const float*)d_in[10];
  const float* bo = (const float*)d_in[11];
  float* out = (float*)d_out;

  // Workspace: 4x bf16 [B,S,D] tensors (8 MB each) + 4x bf16 [D,D] weights (2 MB each) = 40 MB.
  const size_t elems  = (size_t)BB * SS * DD;   // 4M
  const size_t welems = (size_t)DD * DD;        // 1M
  bf16* qbw  = (bf16*)d_ws;
  bf16* kbw  = qbw + elems;
  bf16* vbw  = kbw + elems;
  bf16* ctxb = vbw + elems;
  bf16* Wqb  = ctxb + elems;
  bf16* Wkb  = Wqb + welems;
  bf16* Wvb  = Wkb + welems;
  bf16* Wob  = Wvb + welems;

  cvt_w_kernel<<<dim3(512, 1, 4), 256, 0, stream>>>(Wq, Wk, Wv, Wo, Wqb, Wkb, Wvb, Wob);
  qkv_kernel<<<dim3(256, 1, 3), 256, 0, stream>>>(
      attn_from, attn_to, value, Wqb, bq, Wkb, bk, Wvb, bv, qbw, kbw, vbw);
  attn_kernel<<<dim3(SS / 128, HHN, BB), 256, 0, stream>>>(qbw, kbw, vbw, mask, ctxb);
  proj_kernel<<<dim3(256, 1, 1), 256, 0, stream>>>(ctxb, Wob, bo, out);
}
